// RecurrentGCN_29575144801052
// MI455X (gfx1250) — matrix operand order repaired and checked by reading, not compiled, not run on hardware
//
#include <hip/hip_runtime.h>

typedef __attribute__((ext_vector_type(16))) _Float16 v16h;
typedef __attribute__((ext_vector_type(8)))  float    v8f;

#define F_IN    12
#define H_DIM   32
#define PRE_LEN 12
#define D_CAT   (F_IN + H_DIM)          // 44
#define W_HALF  (D_CAT * H_DIM)         // 1408: offset of second diffusion matrix

__device__ __forceinline__ float fast_sigmoid(float x) {
    return __builtin_amdgcn_rcpf(1.0f + __expf(-x));
}
__device__ __forceinline__ float fast_tanh(float x) {
    // tanh(x) = 2*sigmoid(2x) - 1
    return 2.0f * __builtin_amdgcn_rcpf(1.0f + __expf(-2.0f * x)) - 1.0f;
}

// Fused DCRNN(K=1, H0=0) cell + linear head: per-node
//   z = sigmoid(x @ Az + bz); ht = tanh(x @ Ah + bh)
//   out = relu((1-z)*ht) @ Wlin + blin
// One wave handles a 16-node tile via v_wmma_f32_16x16x32_f16.
__global__ void __launch_bounds__(256)
rgcn_fused_wmma(const float* __restrict__ x,
                const float* __restrict__ Wz,
                const float* __restrict__ bz,
                const float* __restrict__ Wh,
                const float* __restrict__ bh,
                const float* __restrict__ Wlin,
                const float* __restrict__ blin,
                float* __restrict__ out,
                int nNodes, int nTiles)
{
    // Per-wave LDS tile for the D-layout -> A-layout transpose of h.
    // Row stride 34 floats => hi/lo lane halves hit disjoint bank groups.
    __shared__ float hstage[8][16 * 34];

    const int  lane      = threadIdx.x & 31;
    const int  waveInBlk = threadIdx.x >> 5;
    const int  mrow      = lane & 15;
    const bool hiHalf    = lane >= 16;
    const int  klo       = hiHalf ? 16 : 0;   // B-fragment K base for this lane
    const int  mOff      = hiHalf ? 8 : 0;    // C/D-fragment M base for this lane

    // ---- Build B fragments once per wave (weights are tiny & L2-resident) ----
    // B layout (16-bit, 32x16): lane = column (lanes 16-31 repeat cols, K+16 for gates, K+8 for BL);
    // gate element e <-> K = klo + e; BL: K = klo/2 + e, +8 for e >= 8. Rows K >= 12 of the gate weights
    // multiply the zero-padded part of x, so they stay zero.
    v16h Bz0 = {}, Bz1 = {}, Bh0 = {}, Bh1 = {}, BL = {};
    if (!hiHalf) {
        #pragma unroll
        for (int k = 0; k < F_IN; ++k) {
            const int r0 = k * H_DIM + mrow;
            const int r1 = k * H_DIM + 16 + mrow;
            Bz0[k] = (_Float16)(Wz[r0] + Wz[W_HALF + r0]);
            Bz1[k] = (_Float16)(Wz[r1] + Wz[W_HALF + r1]);
            Bh0[k] = (_Float16)(Wh[r0] + Wh[W_HALF + r0]);
            Bh1[k] = (_Float16)(Wh[r1] + Wh[W_HALF + r1]);
        }
    }
    // Output projection: Wlin is [32,12]; K = 32 exactly, N cols 12..15 zero.
    if (mrow < PRE_LEN) {
        #pragma unroll
        for (int e = 0; e < 16; ++e)
            BL[e] = (_Float16)Wlin[((klo >> 1) + e + (e >= 8 ? 8 : 0)) * PRE_LEN + mrow];
    }

    // Per-lane biases (D-layout: N = mrow, +16 for the second tile).
    const float bz0v = bz[mrow], bz1v = bz[mrow + 16];
    const float bh0v = bh[mrow], bh1v = bh[mrow + 16];
    const float blv  = (mrow < PRE_LEN) ? blin[mrow] : 0.0f;

    const int wave = blockIdx.x * (blockDim.x >> 5) + waveInBlk;
    const int nW   = gridDim.x * (blockDim.x >> 5);
    float* myh = &hstage[waveInBlk][0];

    for (int tile = wave; tile < nTiles; tile += nW) {
        const int rowBase = tile * 16;

        // Prefetch the next tile's x rows into cache (global_prefetch_b8).
        const int ptile = tile + nW;
        if (ptile < nTiles)
            __builtin_prefetch(x + (size_t)(ptile * 16 + mrow) * F_IN, 0, 0);

        // ---- Load A fragment: 16x32 f16, K padded 12 -> 32 with zeros ----
        // lo lanes: elems 0..11 = K0..11 ; hi lanes: zero (K16..31).
        int row = rowBase + mrow;
        if (row >= nNodes) row = nNodes - 1;       // clamp: keeps EXEC uniform for WMMA
        const float* xr = x + (size_t)row * F_IN;

        v16h a = {};
        if (!hiHalf) {
            #pragma unroll
            for (int k = 0; k < F_IN; ++k) a[k] = (_Float16)xr[k];
        }

        // ---- Gate GEMMs: 4x v_wmma_f32_16x16x32_f16 ----
        v8f c0 = {};
        v8f zp0 = __builtin_amdgcn_wmma_f32_16x16x32_f16(false, a, false, Bz0, (short)0, c0, false, false);
        v8f zp1 = __builtin_amdgcn_wmma_f32_16x16x32_f16(false, a, false, Bz1, (short)0, c0, false, false);
        v8f hp0 = __builtin_amdgcn_wmma_f32_16x16x32_f16(false, a, false, Bh0, (short)0, c0, false, false);
        v8f hp1 = __builtin_amdgcn_wmma_f32_16x16x32_f16(false, a, false, Bh1, (short)0, c0, false, false);

        // ---- GRU combine (H0=0) + relu; stage h into LDS in (m,n) order ----
        #pragma unroll
        for (int r = 0; r < 8; ++r) {
            float z0 = fast_sigmoid(zp0[r] + bz0v);
            float z1 = fast_sigmoid(zp1[r] + bz1v);
            float t0 = fast_tanh(hp0[r] + bh0v);
            float t1 = fast_tanh(hp1[r] + bh1v);
            float h0 = (1.0f - z0) * t0; h0 = h0 > 0.0f ? h0 : 0.0f;
            float h1 = (1.0f - z1) * t1; h1 = h1 > 0.0f ? h1 : 0.0f;
            myh[(r + mOff) * 34 + mrow]      = h0;   // cols 0..15
            myh[(r + mOff) * 34 + 16 + mrow] = h1;   // cols 16..31
        }

        // ---- Re-gather h in A-layout (same wave -> in-order DS, no barrier) ----
        const int kh = hiHalf ? 8 : 0;
        v16h ah;
        #pragma unroll
        for (int e = 0; e < 8; ++e) {
            ah[e]     = (_Float16)myh[mrow * 34 + kh + e];        // K = kh + e
            ah[8 + e] = (_Float16)myh[mrow * 34 + 16 + kh + e];   // K = 16 + kh + e
        }

        // ---- Output projection: one WMMA, K=32 exact ----
        v8f op = __builtin_amdgcn_wmma_f32_16x16x32_f16(false, ah, false, BL, (short)0, c0, false, false);

        // ---- Store: lanes 0..11 (x2 halves) cover contiguous 48B rows ----
        if (mrow < PRE_LEN) {
            #pragma unroll
            for (int r = 0; r < 8; ++r) {
                const int node = rowBase + r + mOff;
                if (node < nNodes)
                    out[(size_t)node * PRE_LEN + mrow] = op[r] + blv;
            }
        }
    }
}

extern "C" void kernel_launch(void* const* d_in, const int* in_sizes, int n_in,
                              void* d_out, int out_size, void* d_ws, size_t ws_size,
                              hipStream_t stream) {
    // setup_inputs order: 0:x 1:edge_index 2:edge_weight 3:Wz 4:bz 5:Wr 6:br
    //                     7:Wh 8:bh 9:Wlin 10:blin
    // edge_index/edge_weight and the R gate are dead code (H0 == 0).
    const float* x    = (const float*)d_in[0];
    const float* Wz   = (const float*)d_in[3];
    const float* bz   = (const float*)d_in[4];
    const float* Wh   = (const float*)d_in[7];
    const float* bh   = (const float*)d_in[8];
    const float* Wlin = (const float*)d_in[9];
    const float* blin = (const float*)d_in[10];
    float* out = (float*)d_out;

    const int nNodes = in_sizes[0] / F_IN;
    const int nTiles = (nNodes + 15) / 16;

    const int block = 256;                       // 8 waves / block
    const int wavesPerBlock = block / 32;
    int grid = (nTiles + wavesPerBlock - 1) / wavesPerBlock;
    if (grid > 2048) grid = 2048;                // grid-stride amortizes weight setup
    if (grid < 1) grid = 1;

    rgcn_fused_wmma<<<grid, block, 0, stream>>>(x, Wz, bz, Wh, bh, Wlin, blin,
                                                out, nNodes, nTiles);
}
